// SelfAttentionWith2DMask_24687472017811
// MI455X (gfx1250) — matrix operand order repaired and checked by reading, not compiled, not run on hardware
//
#include <hip/hip_runtime.h>
#include <hip/hip_bf16.h>

typedef __attribute__((ext_vector_type(16))) __bf16 v16bf;
typedef __attribute__((ext_vector_type(8)))  float  v8f;

#define B_  128
#define N_  32
#define M_  384
#define D_  256
#define H_  8
#define E_  32
#define O_  256
#define S_  416
#define HE_ 256
#define NEG_INF_ (-1.0e9f)

__device__ __forceinline__ __bf16 f2bf(float f) {
  unsigned u = __builtin_bit_cast(unsigned, f);
  u += 0x7fffu + ((u >> 16) & 1u);                 // round-to-nearest-even
  unsigned short s = (unsigned short)(u >> 16);
  return __builtin_bit_cast(__bf16, s);
}
__device__ __forceinline__ float bf2f(__bf16 h) {
  unsigned short s = __builtin_bit_cast(unsigned short, h);
  unsigned u = ((unsigned)s) << 16;
  return __builtin_bit_cast(float, u);
}

// A fragment (16xK=32 bf16) from row-major src[row][k].
// ISA layout: lanes 0-15 row M=lane, halves 0..7 -> K off..off+7,
// halves 8..15 -> K off+16..off+23, off = 8*(lane>=16).
__device__ __forceinline__ v16bf load_fragA(const __bf16* base, int ld, int row0, int k0, int lane) {
  int r = lane & 15;
  int koff = (lane >> 4) << 3;            // 0 or 8
  const __bf16* p = base + (size_t)(row0 + r) * ld + k0 + koff;
  v16bf a;
#pragma unroll
  for (int i = 0; i < 8; ++i) a[i] = p[i];
#pragma unroll
  for (int i = 0; i < 8; ++i) a[8 + i] = p[16 + i];
  return a;
}

// B fragment (K=32 x 16 bf16) from TRANSPOSED storage src[n][k].
// ISA layout: col N=lane%16, halves 0..7 -> K off..off+7,
// halves 8..15 -> K off+16..off+23, off = 8*(lane>=16) (same as A).
__device__ __forceinline__ v16bf load_fragB(const __bf16* base, int ld, int n0, int k0, int lane) {
  int n = lane & 15;
  int koff = (lane >> 4) << 3;            // 0 or 8
  const __bf16* p = base + (size_t)(n0 + n) * ld + k0 + koff;
  v16bf b;
#pragma unroll
  for (int i = 0; i < 8; ++i) b[i] = p[i];
#pragma unroll
  for (int i = 0; i < 8; ++i) b[8 + i] = p[16 + i];
  return b;
}

__device__ __forceinline__ v8f wmma_bf16(v16bf a, v16bf b, v8f c) {
  return __builtin_amdgcn_wmma_f32_16x16x32_bf16(false, a, false, b, (short)0, c, false, false);
}

// ---------------- kernel 1: weight transposes (f32 -> bf16, [n][k] layouts) ----
__global__ void prep_weights_kernel(const float* __restrict__ qp, const float* __restrict__ kp,
                                    const float* __restrict__ vp, const float* __restrict__ gp,
                                    const float* __restrict__ ow,
                                    __bf16* __restrict__ wT, __bf16* __restrict__ owT) {
  int idx = blockIdx.x * blockDim.x + threadIdx.x;
  const int NW = 4 * H_ * D_ * E_;        // 262144
  if (idx < NW) {
    int w = idx / (H_ * D_ * E_);
    int rem = idx % (H_ * D_ * E_);
    int h = rem / (D_ * E_);
    int de = rem % (D_ * E_);
    int d = de / E_;
    int e = de % E_;
    const float* src = (w == 0) ? qp : (w == 1) ? kp : (w == 2) ? vp : gp;
    wT[(size_t)((w * H_ + h) * E_ + e) * D_ + d] = f2bf(src[(h * D_ + d) * E_ + e]);
  } else {
    int j = idx - NW;
    if (j < O_ * HE_) {
      int k = j / O_;
      int n = j % O_;
      owT[(size_t)n * HE_ + k] = f2bf(ow[k * O_ + n]);
    }
  }
}

// ---------------- kernel 2: LayerNorm(D) -> bf16 xn ; validity flags ----------
__global__ void ln1_valid_kernel(const float* __restrict__ x, const float* __restrict__ pmask,
                                 const float* __restrict__ mmask,
                                 const float* __restrict__ g1, const float* __restrict__ b1,
                                 __bf16* __restrict__ xn, float* __restrict__ valid) {
  int lane = threadIdx.x & 31;
  int wave = threadIdx.x >> 5;
  int t = blockIdx.x * 8 + wave;                    // token index < B_*S_
  const float* row = x + (size_t)t * D_;
  float v[8];
  float s = 0.f, sq = 0.f;
#pragma unroll
  for (int i = 0; i < 8; ++i) { v[i] = row[i * 32 + lane]; s += v[i]; sq += v[i] * v[i]; }
#pragma unroll
  for (int d2 = 1; d2 < 32; d2 <<= 1) { s += __shfl_xor(s, d2); sq += __shfl_xor(sq, d2); }
  float mean = s * (1.f / D_);
  float var = sq * (1.f / D_) - mean * mean;
  float rinv = rsqrtf(var + 1e-6f);
#pragma unroll
  for (int i = 0; i < 8; ++i) {
    int c = i * 32 + lane;
    xn[(size_t)t * D_ + c] = f2bf((v[i] - mean) * rinv * g1[c] + b1[c]);
  }
  int gidx = blockIdx.x * blockDim.x + threadIdx.x;
  if (gidx < B_ * S_) {
    int b = gidx / S_;
    int sI = gidx % S_;
    float tok = (sI < N_) ? pmask[b * N_ + sI] : mmask[b * M_ + (sI - N_)];
    valid[gidx] = (tok == -2.0f) ? 0.f : 1.f;
  }
}

// ---------------- kernel 3: q/k/v/g projections (WMMA GEMM) -------------------
__global__ void __launch_bounds__(128) proj_kernel(const __bf16* __restrict__ xn,
                                                   const __bf16* __restrict__ wT,
                                                   __bf16* __restrict__ Qo, __bf16* __restrict__ Ko,
                                                   __bf16* __restrict__ Vt, __bf16* __restrict__ Go) {
  int wave = threadIdx.x >> 5;                      // 0:q 1:k 2:v 3:g
  int lane = threadIdx.x & 31;
  int bh = blockIdx.x / (S_ / 16);
  int xt = blockIdx.x % (S_ / 16);
  int b = bh / H_;
  int h = bh % H_;
  const __bf16* wbase = wT + (size_t)((wave * H_ + h) * E_) * D_;
  int row0 = b * S_ + xt * 16;
  v8f acc0 = {}; v8f acc1 = {};
  for (int k0 = 0; k0 < D_; k0 += 32) {
    __builtin_prefetch(xn + (size_t)row0 * D_ + k0 + 64, 0, 3);
    v16bf A  = load_fragA(xn, D_, row0, k0, lane);
    v16bf B0 = load_fragB(wbase, D_, 0, k0, lane);
    v16bf B1 = load_fragB(wbase, D_, 16, k0, lane);
    acc0 = wmma_bf16(A, B0, acc0);
    acc1 = wmma_bf16(A, B1, acc1);
  }
  int n = lane & 15;
  int roff = (lane >> 4) << 3;
  size_t base_se = (size_t)bh * S_ * E_;
#pragma unroll
  for (int r = 0; r < 8; ++r) {
    int srow = xt * 16 + r + roff;
    if (wave == 0) {
      Qo[base_se + (size_t)srow * E_ + n]      = f2bf(acc0[r]);
      Qo[base_se + (size_t)srow * E_ + 16 + n] = f2bf(acc1[r]);
    } else if (wave == 1) {
      Ko[base_se + (size_t)srow * E_ + n]      = f2bf(acc0[r]);
      Ko[base_se + (size_t)srow * E_ + 16 + n] = f2bf(acc1[r]);
    } else if (wave == 2) {
      Vt[base_se + (size_t)n * S_ + srow]        = f2bf(acc0[r]);   // V transposed [E][S]
      Vt[base_se + (size_t)(16 + n) * S_ + srow] = f2bf(acc1[r]);
    } else {
      float s0 = 1.f / (1.f + __expf(-acc0[r]));
      float s1 = 1.f / (1.f + __expf(-acc1[r]));
      Go[base_se + (size_t)srow * E_ + n]      = f2bf(s0);
      Go[base_se + (size_t)srow * E_ + 16 + n] = f2bf(s1);
    }
  }
}

// ---------------- kernel 4: fused flash attention with 2D cross mask ----------
__global__ void __launch_bounds__(32) attn_kernel(const __bf16* __restrict__ Qo,
                                                  const __bf16* __restrict__ Ko,
                                                  const __bf16* __restrict__ Vt,
                                                  const __bf16* __restrict__ Go,
                                                  const float* __restrict__ valid,
                                                  __bf16* __restrict__ att_out) {
  int lane = threadIdx.x;
  int bh = blockIdx.x / (S_ / 16);
  int xt = blockIdx.x % (S_ / 16);
  int b = bh / H_;
  int h = bh % H_;
  size_t base_se = (size_t)bh * S_ * E_;
  const __bf16* Qb = Qo + base_se;
  const __bf16* Kb = Ko + base_se;
  const __bf16* Vb = Vt + base_se;
  const float* vld = valid + (size_t)b * S_;
  const float scale = 0.17677669529663687f;         // 1/sqrt(E)

  v16bf Aq = load_fragA(Qb, E_, xt * 16, 0, lane);  // K dim == E == 32: one frag
  int roff = (lane >> 4) << 3;
  int nl = lane & 15;
  float rv[8]; int rg[8]; float m_i[8], l_i[8];
#pragma unroll
  for (int r = 0; r < 8; ++r) {
    int xr = xt * 16 + r + roff;
    rv[r] = vld[xr];
    rg[r] = (xr < N_) ? 1 : 0;
    m_i[r] = -1.0e30f;
    l_i[r] = 0.f;
  }
  v8f acc0 = {}; v8f acc1 = {};
  __shared__ __bf16 P[16 * 32];

  for (int it = 0; it < S_ / 32; ++it) {            // 13 iterations of 32 columns
    int y0 = it * 32;
    v16bf Bk0 = load_fragB(Kb, E_, y0, 0, lane);
    v16bf Bk1 = load_fragB(Kb, E_, y0 + 16, 0, lane);
    v8f z0 = {}; v8f z1 = {};
    v8f s0 = wmma_bf16(Aq, Bk0, z0);
    v8f s1 = wmma_bf16(Aq, Bk1, z1);

    int yc0 = y0 + nl, yc1 = yc0 + 16;
    float cv0 = vld[yc0], cv1 = vld[yc1];
    int cg0 = (yc0 < N_) ? 1 : 0;
    int cg1 = (yc1 < N_) ? 1 : 0;
#pragma unroll
    for (int r = 0; r < 8; ++r) {
      bool ok0 = (rg[r] != cg0) && (rv[r] > 0.5f) && (cv0 > 0.5f);
      bool ok1 = (rg[r] != cg1) && (rv[r] > 0.5f) && (cv1 > 0.5f);
      float sc0 = s0[r] * scale + (ok0 ? 0.f : NEG_INF_);
      float sc1 = s1[r] * scale + (ok1 ? 0.f : NEG_INF_);
      float t = fmaxf(sc0, sc1);
#pragma unroll
      for (int d2 = 1; d2 < 16; d2 <<= 1) t = fmaxf(t, __shfl_xor(t, d2));
      float newm = fmaxf(m_i[r], t);
      float alpha = __expf(m_i[r] - newm);
      m_i[r] = newm;
      float p0 = ok0 ? __expf(sc0 - newm) : 0.f;    // post-softmax mask == zero P
      float p1 = ok1 ? __expf(sc1 - newm) : 0.f;
      float rs = p0 + p1;
#pragma unroll
      for (int d2 = 1; d2 < 16; d2 <<= 1) rs += __shfl_xor(rs, d2);
      l_i[r] = l_i[r] * alpha + rs;
      acc0[r] *= alpha;
      acc1[r] *= alpha;
      P[(r + roff) * 32 + nl]      = f2bf(p0);
      P[(r + roff) * 32 + 16 + nl] = f2bf(p1);
    }
    __syncthreads();
    v16bf Ap  = load_fragA(P, 32, 0, 0, lane);
    v16bf Bv0 = load_fragB(Vb, S_, 0, y0, lane);
    v16bf Bv1 = load_fragB(Vb, S_, 16, y0, lane);
    acc0 = wmma_bf16(Ap, Bv0, acc0);
    acc1 = wmma_bf16(Ap, Bv1, acc1);
    __syncthreads();
  }
#pragma unroll
  for (int r = 0; r < 8; ++r) {
    int xr = xt * 16 + r + roff;
    float inv = 1.f / fmaxf(l_i[r], 1e-20f);
    float g0 = bf2f(Go[base_se + (size_t)xr * E_ + nl]);
    float g1 = bf2f(Go[base_se + (size_t)xr * E_ + 16 + nl]);
    size_t orow = (size_t)(b * S_ + xr) * HE_ + h * E_;
    att_out[orow + nl]      = f2bf(acc0[r] * inv * g0);
    att_out[orow + 16 + nl] = f2bf(acc1[r] * inv * g1);
  }
}

// ---------------- kernel 5: output projection + bias + LayerNorm(O) -----------
__global__ void __launch_bounds__(256) outproj_ln_kernel(const __bf16* __restrict__ att_out,
                                                         const __bf16* __restrict__ owT,
                                                         const float* __restrict__ bias,
                                                         const float* __restrict__ g2,
                                                         const float* __restrict__ b2,
                                                         float* __restrict__ out) {
  int wave = threadIdx.x >> 5;
  int lane = threadIdx.x & 31;
  int row0 = blockIdx.x * 16;
  int n0 = wave * 32;
  int nl = lane & 15;
  int roff = (lane >> 4) << 3;
  float bias0 = bias[n0 + nl];
  float bias1 = bias[n0 + 16 + nl];
  v8f acc0 = {}; v8f acc1 = {};
  for (int k0 = 0; k0 < HE_; k0 += 32) {
    __builtin_prefetch(att_out + (size_t)row0 * HE_ + k0 + 64, 0, 3);
    v16bf A  = load_fragA(att_out, HE_, row0, k0, lane);
    v16bf B0 = load_fragB(owT, HE_, n0, k0, lane);
    v16bf B1 = load_fragB(owT, HE_, n0 + 16, k0, lane);
    acc0 = wmma_bf16(A, B0, acc0);
    acc1 = wmma_bf16(A, B1, acc1);
  }
  __shared__ float ssum[16][8];
  __shared__ float ssq[16][8];
  float v0[8], v1[8];
#pragma unroll
  for (int r = 0; r < 8; ++r) {
    v0[r] = acc0[r] + bias0;
    v1[r] = acc1[r] + bias1;
    float ps = v0[r] + v1[r];
    float pq = v0[r] * v0[r] + v1[r] * v1[r];
#pragma unroll
    for (int d2 = 1; d2 < 16; d2 <<= 1) { ps += __shfl_xor(ps, d2); pq += __shfl_xor(pq, d2); }
    if (nl == 0) { ssum[r + roff][wave] = ps; ssq[r + roff][wave] = pq; }
  }
  __syncthreads();
  float gc0 = g2[n0 + nl], bc0 = b2[n0 + nl];
  float gc1 = g2[n0 + 16 + nl], bc1 = b2[n0 + 16 + nl];
#pragma unroll
  for (int r = 0; r < 8; ++r) {
    int row = r + roff;
    float s = 0.f, q = 0.f;
#pragma unroll
    for (int w = 0; w < 8; ++w) { s += ssum[row][w]; q += ssq[row][w]; }
    float mean = s * (1.f / O_);
    float var = q * (1.f / O_) - mean * mean;
    float rinv = rsqrtf(var + 1e-6f);
    size_t orow = (size_t)(row0 + row) * O_;
    out[orow + n0 + nl]      = (v0[r] - mean) * rinv * gc0 + bc0;
    out[orow + n0 + 16 + nl] = (v1[r] - mean) * rinv * gc1 + bc1;
  }
}

// ---------------- launch -------------------------------------------------------
extern "C" void kernel_launch(void* const* d_in, const int* in_sizes, int n_in,
                              void* d_out, int out_size, void* d_ws, size_t ws_size,
                              hipStream_t stream) {
  (void)in_sizes; (void)n_in; (void)out_size; (void)ws_size;
  const float* x     = (const float*)d_in[0];
  const float* pmask = (const float*)d_in[1];
  const float* mmask = (const float*)d_in[2];
  const float* qp    = (const float*)d_in[3];
  const float* kp    = (const float*)d_in[4];
  const float* vp    = (const float*)d_in[5];
  const float* gp    = (const float*)d_in[6];
  const float* ow    = (const float*)d_in[7];
  const float* ob    = (const float*)d_in[8];
  const float* g1    = (const float*)d_in[9];
  const float* b1    = (const float*)d_in[10];
  const float* g2    = (const float*)d_in[11];
  const float* b2    = (const float*)d_in[12];
  float* out = (float*)d_out;

  char* ws = (char*)d_ws;
  size_t off = 0;
  auto alloc = [&](size_t bytes) -> void* {
    void* p = ws + off;
    off += (bytes + 255) & ~(size_t)255;
    return p;
  };
  __bf16* xn  = (__bf16*)alloc((size_t)B_ * S_ * D_ * 2);       // 27.3 MB
  float*  vld = (float*) alloc((size_t)B_ * S_ * 4);            // 0.2 MB
  __bf16* wT  = (__bf16*)alloc((size_t)4 * H_ * E_ * D_ * 2);   // 0.5 MB
  __bf16* owT = (__bf16*)alloc((size_t)O_ * HE_ * 2);           // 0.13 MB
  __bf16* Qm  = (__bf16*)alloc((size_t)B_ * H_ * S_ * E_ * 2);  // 27.3 MB
  __bf16* Km  = (__bf16*)alloc((size_t)B_ * H_ * S_ * E_ * 2);  // 27.3 MB
  __bf16* Vtm = (__bf16*)alloc((size_t)B_ * H_ * S_ * E_ * 2);  // 27.3 MB
  __bf16* Gm  = (__bf16*)alloc((size_t)B_ * H_ * S_ * E_ * 2);  // 27.3 MB
  __bf16* att = (__bf16*)alloc((size_t)B_ * S_ * HE_ * 2);      // 27.3 MB  (~164 MB total, fits L2)

  prep_weights_kernel<<<(4 * H_ * D_ * E_ + O_ * HE_ + 255) / 256, 256, 0, stream>>>(
      qp, kp, vp, gp, ow, wT, owT);
  ln1_valid_kernel<<<(B_ * S_) / 8, 256, 0, stream>>>(x, pmask, mmask, g1, b1, xn, vld);
  proj_kernel<<<B_ * H_ * (S_ / 16), 128, 0, stream>>>(xn, wT, Qm, Km, Vtm, Gm);
  attn_kernel<<<B_ * H_ * (S_ / 16), 32, 0, stream>>>(Qm, Km, Vtm, Gm, vld, att);
  outproj_ln_kernel<<<(B_ * S_) / 16, 256, 0, stream>>>(att, owT, ob, g2, b2, out);
}
